// MultiHeadAttention_64742337020451
// MI455X (gfx1250) — matrix operand order repaired and checked by reading, not compiled, not run on hardware
//
#include <hip/hip_runtime.h>
#include <hip/hip_bf16.h>

// MHA forward for B=2, S=2048, D=1024, H=16, HD=64 on gfx1250 (wave32, WMMA).
// Pipeline: f32->f16 convert -> proj(Q,K,V) -> flash attention -> out proj.
// All matmuls via v_wmma_f32_16x16x32_f16 (f16 operands, f32 accumulate).

#define BB   2
#define SS   2048
#define DDIM 1024
#define HH   16
#define HDIM 64
#define MM   (BB * SS)  // 4096 rows total

typedef _Float16 half_t;
typedef __attribute__((ext_vector_type(16))) _Float16 v16h;
typedef __attribute__((ext_vector_type(8)))  _Float16 v8h;
typedef __attribute__((ext_vector_type(8)))  float    v8f;

__device__ __forceinline__ v8f wmma_f16(v16h a, v16h b, v8f c) {
  // v_wmma_f32_16x16x32_f16  (neg_a, A, neg_b, B, c_mod, C, reuse_a, reuse_b)
  return __builtin_amdgcn_wmma_f32_16x16x32_f16(false, a, false, b, (short)0, c,
                                                false, false);
}

// ---- Operand loaders (f16 memory) ------------------------------------------
// A-operand (16x32): lane holds row m=lane&15; per ISA table:
//   e<8 : K = e + 8g   (g = lane>>4);   e>=8: K = (e-8) + 8g + 16
// => two contiguous 16B loads at column offsets 8g and 16+8g.
__device__ __forceinline__ v16h load_a16(const half_t* __restrict__ base,
                                         int ld, int lane) {
  int m = lane & 15, g = lane >> 4;
  const half_t* p = base + (size_t)m * ld + 8 * g;
  v8h lo = *(const v8h*)(p);
  v8h hi = *(const v8h*)(p + 16);
  v16h a;
#pragma unroll
  for (int e = 0; e < 8; ++e) { a[e] = lo[e]; a[e + 8] = hi[e]; }
  return a;
}

// B-operand (32x16): lane holds column n=lane&15; same K order as A (e<8: e+8g, e>=8: e-8+8g+16).
// `base` = top-left of a row-major array whose ROWS are the B-matrix COLUMNS.
__device__ __forceinline__ v16h load_b16(const half_t* __restrict__ base,
                                         int ld, int lane) {
  int n = lane & 15, g = lane >> 4;
  const half_t* p = base + (size_t)n * ld + 8 * g;
  v8h lo = *(const v8h*)(p);
  v8h hi = *(const v8h*)(p + 16);
  v16h b;
#pragma unroll
  for (int e = 0; e < 8; ++e) { b[e] = lo[e]; b[e + 8] = hi[e]; }
  return b;
}

// ---- Kernel 0: f32 -> f16 bulk convert (bandwidth bound, one pass) ---------
__global__ __launch_bounds__(256) void cvt_kernel(
    const float* __restrict__ src, half_t* __restrict__ dst, int n8) {
  int i = blockIdx.x * blockDim.x + threadIdx.x;
  if (i >= n8) return;
  float4 f0 = ((const float4*)src)[2 * i];
  float4 f1 = ((const float4*)src)[2 * i + 1];
  v8h o;
  o[0] = (half_t)f0.x; o[1] = (half_t)f0.y;
  o[2] = (half_t)f0.z; o[3] = (half_t)f0.w;
  o[4] = (half_t)f1.x; o[5] = (half_t)f1.y;
  o[6] = (half_t)f1.z; o[7] = (half_t)f1.w;
  ((v8h*)dst)[i] = o;
}

// ---- Kernel 1: P = (X @ W^T + bias) * scale, f16 in -> f16 out -------------
// One wave computes a 32x64 output tile (2 A-tiles x 4 B-tiles = 8 WMMAs/k0).
// Blocks of 8 waves share the column group => B tiles hit L0 hot.
// transposed==1 stores per-batch transposed [B][D][S] (for V).
__global__ __launch_bounds__(256) void proj_kernel(
    const half_t* __restrict__ X, const half_t* __restrict__ W,
    const float* __restrict__ bias, half_t* __restrict__ out,
    int transposed, float scale) {
  int wave = (blockIdx.x * blockDim.x + threadIdx.x) >> 5;
  int lane = threadIdx.x & 31;
  int rowTile = wave & 127;      // M/32 = 128 row tiles (fast within block)
  int colGrp  = wave >> 7;       // D/64 = 16 column groups
  int r0 = rowTile << 5;
  int n0 = colGrp << 6;
  int n = lane & 15, g = lane >> 4;

  v8f accA[4] = {}, accB[4] = {};
  for (int k0 = 0; k0 < DDIM; k0 += 32) {
    v16h a0 = load_a16(X + (size_t)r0 * DDIM + k0, DDIM, lane);
    v16h a1 = load_a16(X + (size_t)(r0 + 16) * DDIM + k0, DDIM, lane);
#pragma unroll
    for (int t = 0; t < 4; ++t) {
      v16h b = load_b16(W + (size_t)(n0 + 16 * t) * DDIM + k0, DDIM, lane);
      accA[t] = wmma_f16(a0, b, accA[t]);
      accB[t] = wmma_f16(a1, b, accB[t]);
    }
  }

  if (transposed) {
    int bidx = r0 / SS;
    int s0 = (r0 % SS) + 8 * g;
#pragma unroll
    for (int t = 0; t < 4; ++t) {
      int col = n0 + 16 * t + n;
      float bv = bias[col];
      v8h pkA, pkB;
#pragma unroll
      for (int r = 0; r < 8; ++r) {
        pkA[r] = (half_t)((accA[t][r] + bv) * scale);
        pkB[r] = (half_t)((accB[t][r] + bv) * scale);
      }
      half_t* o = out + (size_t)bidx * DDIM * SS + (size_t)col * SS + s0;
      *(v8h*)(o)      = pkA;
      *(v8h*)(o + 16) = pkB;
    }
  } else {
#pragma unroll
    for (int t = 0; t < 4; ++t) {
      int col = n0 + 16 * t + n;
      float bv = bias[col];
#pragma unroll
      for (int r = 0; r < 8; ++r) {
        out[(size_t)(r0 + r + 8 * g) * DDIM + col] =
            (half_t)((accA[t][r] + bv) * scale);
        out[(size_t)(r0 + 16 + r + 8 * g) * DDIM + col] =
            (half_t)((accB[t][r] + bv) * scale);
      }
    }
  }
}

// ---- Kernel 2: causal flash attention --------------------------------------
// One wave handles 16 query rows of one (b,h). Scores computed transposed:
// S^T = K * Q^T  => C-layout lane = query index, so softmax state is
// lane-local (one shfl_xor(16) merges the half-wave row split), and the
// S^T C-layout maps directly onto the A-operand layout of ctx = P * V.
__global__ __launch_bounds__(256) void attn_kernel(
    const half_t* __restrict__ QP, const half_t* __restrict__ KP,
    const half_t* __restrict__ VPt, half_t* __restrict__ CTX) {
  int wave = (blockIdx.x * blockDim.x + threadIdx.x) >> 5;
  int lane = threadIdx.x & 31;
  int qt = wave & 127;            // S/16 (fast within block: shares K/V)
  int h  = (wave >> 7) & 15;
  int b  = wave >> 11;
  int q0 = qt << 4;
  int d0 = h << 6;
  int n = lane & 15, g = lane >> 4;
  int qg = q0 + n;                // this lane's global query row

  // Q B-operands for both 32-wide head-dim chunks: invariant over K loop.
  const half_t* qbase = QP + (size_t)(b * SS + q0) * DDIM + d0;
  v16h bq0 = load_b16(qbase,      DDIM, lane);
  v16h bq1 = load_b16(qbase + 32, DDIM, lane);

  const half_t* kbase = KP + (size_t)(b * SS) * DDIM + d0;
  const half_t* vbase = VPt + (size_t)b * DDIM * SS + (size_t)d0 * SS;

  v8f O0 = {}, O1 = {}, O2 = {}, O3 = {};
  float mrow = -3.0e38f, lsum = 0.0f;

  for (int kb = 0; kb < q0 + 16; kb += 32) {
    if (kb + 48 < q0 + 16) {  // prefetch next K block (global_prefetch_b8)
      __builtin_prefetch(kbase + (size_t)(kb + 32 + n) * DDIM + 8 * g, 0, 0);
      __builtin_prefetch(kbase + (size_t)(kb + 48 + n) * DDIM + 8 * g, 0, 0);
    }
    // S^T tiles: t0 = k rows [kb,kb+16), t1 = [kb+16,kb+32); contract HD=64.
    v8f st0 = {}, st1 = {};
    st0 = wmma_f16(load_a16(kbase + (size_t)kb * DDIM,        DDIM, lane), bq0, st0);
    st0 = wmma_f16(load_a16(kbase + (size_t)kb * DDIM + 32,   DDIM, lane), bq1, st0);
    st1 = wmma_f16(load_a16(kbase + (size_t)(kb + 16) * DDIM,      DDIM, lane), bq0, st1);
    st1 = wmma_f16(load_a16(kbase + (size_t)(kb + 16) * DDIM + 32, DDIM, lane), bq1, st1);

    if (kb + 31 > q0) {  // causal mask only touches the diagonal block(s)
#pragma unroll
      for (int r = 0; r < 8; ++r) {
        if (kb + r + 8 * g > qg)      st0[r] = -1.0e9f;
        if (kb + 16 + r + 8 * g > qg) st1[r] = -1.0e9f;
      }
    }

    // online softmax; lane holds query q=n, k-rows (r+8g) and (16+r+8g)
    float bmax = st0[0];
#pragma unroll
    for (int r = 0; r < 8; ++r) {
      bmax = fmaxf(bmax, st0[r]);
      bmax = fmaxf(bmax, st1[r]);
    }
    bmax = fmaxf(bmax, __shfl_xor(bmax, 16, 32));
    float mnew = fmaxf(mrow, bmax);
    float alpha = __expf(mrow - mnew);
    mrow = mnew;

    float rsum = 0.0f;
    v16h pa;  // A-operand for P*V: pa[e] = st_{e>>3}[e&7] (pure reg permute)
#pragma unroll
    for (int r = 0; r < 8; ++r) {
      float e0 = __expf(st0[r] - mnew);
      float e1 = __expf(st1[r] - mnew);
      rsum += e0 + e1;
      pa[r]     = (half_t)e0;
      pa[r + 8] = (half_t)e1;
    }
    rsum += __shfl_xor(rsum, 16, 32);
    lsum = lsum * alpha + rsum;

    // rescale accumulators: O rows are q = r + 8g; alpha lives in lane (q&15)
#pragma unroll
    for (int r = 0; r < 8; ++r) {
      float ar = __shfl(alpha, r + 8 * g, 32);
      O0[r] *= ar; O1[r] *= ar; O2[r] *= ar; O3[r] *= ar;
    }
    // ctx += P(16x32) @ V(32x64); V B-operand contiguous thanks to VPt layout
    O0 = wmma_f16(pa, load_b16(vbase + (size_t)0  * SS + kb, SS, lane), O0);
    O1 = wmma_f16(pa, load_b16(vbase + (size_t)16 * SS + kb, SS, lane), O1);
    O2 = wmma_f16(pa, load_b16(vbase + (size_t)32 * SS + kb, SS, lane), O2);
    O3 = wmma_f16(pa, load_b16(vbase + (size_t)48 * SS + kb, SS, lane), O3);
  }

  // normalize by row sums and store ctx (row-major [B*S, D], f16)
#pragma unroll
  for (int r = 0; r < 8; ++r) {
    float inv = 1.0f / __shfl(lsum, r + 8 * g, 32);
    O0[r] *= inv; O1[r] *= inv; O2[r] *= inv; O3[r] *= inv;
  }
#pragma unroll
  for (int r = 0; r < 8; ++r) {
    size_t row = (size_t)(b * SS + q0 + r + 8 * g) * DDIM + d0 + n;
    CTX[row]      = (half_t)O0[r];
    CTX[row + 16] = (half_t)O1[r];
    CTX[row + 32] = (half_t)O2[r];
    CTX[row + 48] = (half_t)O3[r];
  }
}

// ---- Kernel 3: out = CTX @ Wo^T + b_o, f16 in -> f32 out -------------------
__global__ __launch_bounds__(256) void outproj_kernel(
    const half_t* __restrict__ CTX, const half_t* __restrict__ W,
    const float* __restrict__ bias, float* __restrict__ out) {
  int wave = (blockIdx.x * blockDim.x + threadIdx.x) >> 5;
  int lane = threadIdx.x & 31;
  int rowTile = wave & 127;
  int colGrp  = wave >> 7;
  int r0 = rowTile << 5;
  int n0 = colGrp << 6;
  int n = lane & 15, g = lane >> 4;

  v8f accA[4] = {}, accB[4] = {};
  for (int k0 = 0; k0 < DDIM; k0 += 32) {
    v16h a0 = load_a16(CTX + (size_t)r0 * DDIM + k0, DDIM, lane);
    v16h a1 = load_a16(CTX + (size_t)(r0 + 16) * DDIM + k0, DDIM, lane);
#pragma unroll
    for (int t = 0; t < 4; ++t) {
      v16h b = load_b16(W + (size_t)(n0 + 16 * t) * DDIM + k0, DDIM, lane);
      accA[t] = wmma_f16(a0, b, accA[t]);
      accB[t] = wmma_f16(a1, b, accB[t]);
    }
  }
#pragma unroll
  for (int t = 0; t < 4; ++t) {
    int col = n0 + 16 * t + n;
    float bv = bias[col];
#pragma unroll
    for (int r = 0; r < 8; ++r) {
      out[(size_t)(r0 + r + 8 * g) * DDIM + col]      = accA[t][r] + bv;
      out[(size_t)(r0 + 16 + r + 8 * g) * DDIM + col] = accB[t][r] + bv;
    }
  }
}

// ---- Host launcher ---------------------------------------------------------
extern "C" void kernel_launch(void* const* d_in, const int* in_sizes, int n_in,
                              void* d_out, int out_size, void* d_ws,
                              size_t ws_size, hipStream_t stream) {
  (void)in_sizes; (void)n_in; (void)out_size; (void)ws_size;
  const float* q   = (const float*)d_in[0];
  const float* k   = (const float*)d_in[1];
  const float* v   = (const float*)d_in[2];
  // d_in[3] = causal mask (ignored; mask computed analytically)
  const float* w_q = (const float*)d_in[4];
  const float* b_q = (const float*)d_in[5];
  const float* w_k = (const float*)d_in[6];
  const float* b_k = (const float*)d_in[7];
  const float* w_v = (const float*)d_in[8];
  const float* b_v = (const float*)d_in[9];
  const float* w_o = (const float*)d_in[10];
  const float* b_o = (const float*)d_in[11];

  const size_t XN = (size_t)MM * DDIM;    // 4M elements
  const size_t WN = (size_t)DDIM * DDIM;  // 1M elements

  half_t* X16q = (half_t*)d_ws;           // f16 copies of inputs
  half_t* X16k = X16q + XN;
  half_t* X16v = X16k + XN;
  half_t* W16q = X16v + XN;               // f16 copies of weights
  half_t* W16k = W16q + WN;
  half_t* W16v = W16k + WN;
  half_t* W16o = W16v + WN;
  half_t* QP   = W16o + WN;               // [B*S, D] scaled by 1/8
  half_t* KP   = QP + XN;                 // [B*S, D]
  half_t* VPt  = KP + XN;                 // [B][D][S] (transposed V)
  half_t* CTX  = VPt + XN;                // [B*S, D]

  dim3 blk(256);
  // one-pass f32 -> f16 conversions (bandwidth bound)
  cvt_kernel<<<dim3((int)(XN / 8 / 256)), blk, 0, stream>>>(q, X16q, (int)(XN / 8));
  cvt_kernel<<<dim3((int)(XN / 8 / 256)), blk, 0, stream>>>(k, X16k, (int)(XN / 8));
  cvt_kernel<<<dim3((int)(XN / 8 / 256)), blk, 0, stream>>>(v, X16v, (int)(XN / 8));
  cvt_kernel<<<dim3((int)(WN / 8 / 256)), blk, 0, stream>>>(w_q, W16q, (int)(WN / 8));
  cvt_kernel<<<dim3((int)(WN / 8 / 256)), blk, 0, stream>>>(w_k, W16k, (int)(WN / 8));
  cvt_kernel<<<dim3((int)(WN / 8 / 256)), blk, 0, stream>>>(w_v, W16v, (int)(WN / 8));
  cvt_kernel<<<dim3((int)(WN / 8 / 256)), blk, 0, stream>>>(w_o, W16o, (int)(WN / 8));

  // projections: 2048 waves = (M/32 row tiles) x (D/64 col groups)
  dim3 gproj(256);
  proj_kernel<<<gproj, blk, 0, stream>>>(X16q, W16q, b_q, QP, 0, 0.125f);
  proj_kernel<<<gproj, blk, 0, stream>>>(X16k, W16k, b_k, KP, 0, 1.0f);
  proj_kernel<<<gproj, blk, 0, stream>>>(X16v, W16v, b_v, VPt, 1, 1.0f);

  // attention: 4096 waves = B * H * (S/16)
  attn_kernel<<<dim3(512), blk, 0, stream>>>(QP, KP, VPt, CTX);

  // output projection
  outproj_kernel<<<gproj, blk, 0, stream>>>(CTX, W16o, b_o, (float*)d_out);
}
